// EngramModule_11914239279775
// MI455X (gfx1250) — matrix operand order repaired and checked by reading, not compiled, not run on hardware
//
#include <hip/hip_runtime.h>
#include <hip/hip_bf16.h>

typedef __attribute__((ext_vector_type(16))) _Float16 v16h;
typedef __attribute__((ext_vector_type(8)))  float    v8f;

union Frag16 { v16h v; unsigned int u[8]; };

#define HIDDEN   2048
#define NCOLS    4096           // [Wk out | Wv out]
#define SEQ      4096
#define LDSTRIDE 34             // 32 halfs + 2 pad (68B row -> 17 banks, coprime with 64)

// ---------------------------------------------------------------------------
// Kernel 1: fused hash + table-gather + dual GEMM (memory@Wk.T | memory@Wv.T)
// 128 threads = 4 waves. Tile: M=128 positions, N=64 output cols. K=2048.
// Per wave: 32(M)x64(N) -> 2 A frags x 4 B frags = 8 wmma per 32-K step.
// f16 WMMA (v_wmma_f32_16x16x32_f16), f32 accumulate, results -> ws (mkv).
// ---------------------------------------------------------------------------
__global__ __launch_bounds__(128)
void engram_gemm_kernel(const long long* __restrict__ ids,
                        const float* __restrict__ tables,
                        const float* __restrict__ Wk,
                        const float* __restrict__ Wv,
                        const long long* __restrict__ hm2,
                        const long long* __restrict__ ho2,
                        const long long* __restrict__ hm3,
                        const long long* __restrict__ ho3,
                        float* __restrict__ mkv)
{
    __shared__ unsigned int sHash[128][8];
    __shared__ __align__(16) _Float16 sA[128 * LDSTRIDE];
    __shared__ __align__(16) _Float16 sB[64 * LDSTRIDE];

    const int tid   = threadIdx.x;
    const int ntile = blockIdx.x & 63;        // 64 tiles across N=4096
    const int mtile = blockIdx.x >> 6;        // 128 tiles across M=16384
    const int m0    = mtile * 128;
    const int n0    = ntile * 64;

    // ---- per-tile hash ids (8 heads x 128 positions, one position/thread) ----
    {
        int pos = m0 + tid;
        int si  = pos & (SEQ - 1);
        const long long* row = ids + ((size_t)(pos >> 12) << 12);
        unsigned long long t0 = (unsigned long long)row[si];
        unsigned long long t1 = (si >= 1) ? (unsigned long long)row[si - 1] : 0ull;
        unsigned long long t2 = (si >= 2) ? (unsigned long long)row[si - 2] : 0ull;
#pragma unroll
        for (int h = 0; h < 4; ++h) {
            unsigned long long mix = (t1 * (unsigned long long)hm2[h * 2 + 0]) ^
                                     (t0 * (unsigned long long)hm2[h * 2 + 1]);
            unsigned long long hv  = (mix + (unsigned long long)ho2[h]) % 65535ull + 1ull;
            sHash[tid][h] = (si >= 1) ? (unsigned int)hv : 0u;
        }
#pragma unroll
        for (int h = 0; h < 4; ++h) {
            unsigned long long mix = (t2 * (unsigned long long)hm3[h * 3 + 0]) ^
                                     (t1 * (unsigned long long)hm3[h * 3 + 1]) ^
                                     (t0 * (unsigned long long)hm3[h * 3 + 2]);
            unsigned long long hv  = (mix + (unsigned long long)ho3[h]) % 65535ull + 1ull;
            sHash[tid][4 + h] = (si >= 2) ? (unsigned int)hv : 0u;
        }
    }

    const float* Wsel = (n0 < HIDDEN) ? Wk : Wv;
    const int    nc   = n0 & (HIDDEN - 1);

    const int wave = tid >> 5;
    const int lane = tid & 31;
    const int Lm   = lane & 15;
    const int half = lane >> 4;

    const int brow = tid >> 1;            // 0..63: B row staged by this thread
    const int bcol = (tid & 1) * 16;      // half-row: cols [bcol, bcol+16)

    v8f acc[2][4];
#pragma unroll
    for (int a = 0; a < 2; ++a)
#pragma unroll
        for (int nb = 0; nb < 4; ++nb) acc[a][nb] = (v8f){};

    for (int k0 = 0; k0 < HIDDEN; k0 += 32) {
        __syncthreads();
        // ---- stage A: one gathered table row per thread, f32 -> f16 ----
        {
            const int head = k0 >> 8;          // 256 features per head
            const int d0   = (k0 & 255);
            const unsigned int r = sHash[tid][head];
            const float* src = tables + (((size_t)head << 16) + r) * 256 + d0;
            _Float16* dst = &sA[tid * LDSTRIDE];
#pragma unroll
            for (int i = 0; i < 32; i += 4) {
                float4 f = *(const float4*)(src + i);
                dst[i + 0] = (_Float16)f.x; dst[i + 1] = (_Float16)f.y;
                dst[i + 2] = (_Float16)f.z; dst[i + 3] = (_Float16)f.w;
            }
        }
        // ---- stage B: Wk/Wv rows, f32 -> f16 (half-row per thread) ----
        {
            const float* src = Wsel + (size_t)(nc + brow) * HIDDEN + k0 + bcol;
            _Float16* dst = &sB[brow * LDSTRIDE + bcol];
#pragma unroll
            for (int i = 0; i < 16; i += 4) {
                float4 f = *(const float4*)(src + i);
                dst[i + 0] = (_Float16)f.x; dst[i + 1] = (_Float16)f.y;
                dst[i + 2] = (_Float16)f.z; dst[i + 3] = (_Float16)f.w;
            }
        }
        __syncthreads();

        // ---- preload 2 A fragments (ISA 16-bit A 16x32 layout) ----
        Frag16 af[2];
#pragma unroll
        for (int a = 0; a < 2; ++a) {
            const _Float16* Ar = &sA[(wave * 32 + a * 16 + Lm) * LDSTRIDE];
#pragma unroll
            for (int p = 0; p < 8; ++p) {
                const int K0 = 2 * p + 8 * half + ((p >= 4) ? 8 : 0);
                af[a].u[p] = *(const unsigned int*)(Ar + K0);
            }
        }
        // ---- preload 4 B fragments (ISA 16-bit B 32x16 layout) ----
        Frag16 bf[4];
#pragma unroll
        for (int nb = 0; nb < 4; ++nb) {
            const _Float16* Br = &sB[(nb * 16 + Lm) * LDSTRIDE];
#pragma unroll
            for (int p = 0; p < 8; ++p) {
                const int K0 = 2 * p + 8 * half + ((p >= 4) ? 8 : 0);
                bf[nb].u[p] = *(const unsigned int*)(Br + K0);
            }
        }
        // ---- 8 WMMAs back-to-back ----
#pragma unroll
        for (int nb = 0; nb < 4; ++nb) {
#pragma unroll
            for (int a = 0; a < 2; ++a) {
                acc[a][nb] = __builtin_amdgcn_wmma_f32_16x16x32_f16(
                    false, af[a].v, false, bf[nb].v, (short)0, acc[a][nb], false, false);
            }
        }
    }

    // ---- store D (VGPR g: M=g+8*half, N=Lm) ----
#pragma unroll
    for (int a = 0; a < 2; ++a) {
#pragma unroll
        for (int nb = 0; nb < 4; ++nb) {
            const int col = n0 + nb * 16 + Lm;
            size_t base = (size_t)(m0 + wave * 32 + a * 16 + 8 * half) * NCOLS + col;
#pragma unroll
            for (int g = 0; g < 8; ++g)
                mkv[base + (size_t)g * NCOLS] = acc[a][nb][g];
        }
    }
}

// ---------------------------------------------------------------------------
// Kernel 2: per-position RMS-norm of mk/mv, gate dot + sigmoid, write gated
// in place over the mk half of the workspace row.
// ---------------------------------------------------------------------------
__global__ __launch_bounds__(256)
void engram_gate_kernel(const float* __restrict__ hidden,
                        const float* __restrict__ knw,
                        const float* __restrict__ vnw,
                        float* __restrict__ mkv)
{
    const int pos = blockIdx.x;
    const int tid = threadIdx.x;
    float* rowp = mkv + (size_t)pos * NCOLS;
    const float* hp = hidden + (size_t)pos * HIDDEN;

    float mvl[8], vwl[8];
    float ssk = 0.f, ssv = 0.f, dk = 0.f;
#pragma unroll
    for (int i = 0; i < 8; ++i) {
        const int c = tid + i * 256;
        const float a = rowp[c];
        const float v = rowp[HIDDEN + c];
        const float h = hp[c];
        mvl[i] = v; vwl[i] = vnw[c];
        ssk += a * a;
        ssv += v * v;
        dk  += h * knw[c] * a;
    }

    __shared__ float sred[3 * 256];
    sred[tid] = ssk; sred[256 + tid] = ssv; sred[512 + tid] = dk;
    __syncthreads();
    for (int off = 128; off > 0; off >>= 1) {
        if (tid < off) {
            sred[tid]       += sred[tid + off];
            sred[256 + tid] += sred[256 + tid + off];
            sred[512 + tid] += sred[512 + tid + off];
        }
        __syncthreads();
    }
    const float rsk = rsqrtf(sred[0]   * (1.0f / HIDDEN) + 1e-6f);
    const float rsv = rsqrtf(sred[256] * (1.0f / HIDDEN) + 1e-6f);
    const float logit = sred[512] * rsk * 0.022097086912079608f - 4.0f; // 1/sqrt(2048)
    const float gate  = 1.0f / (1.0f + expf(-logit));

#pragma unroll
    for (int i = 0; i < 8; ++i) {
        const int c = tid + i * 256;
        rowp[c] = gate * vwl[i] * mvl[i] * rsv;   // gated, in place
    }
}

// ---------------------------------------------------------------------------
// Kernel 3: causal depthwise conv (K=3) + residual.
// out[s] = g[s] + w2*g[s] + w1*g[s-1] + w0*g[s-2]   (within each batch row)
// ---------------------------------------------------------------------------
__global__ __launch_bounds__(256)
void engram_conv_kernel(const float* __restrict__ gated,   // ws rows, stride NCOLS
                        const float* __restrict__ conv_w,  // [HIDDEN][3]
                        float* __restrict__ out)
{
    const int pos = blockIdx.x;
    const int si  = pos & (SEQ - 1);
    const float* g0 = gated + (size_t)pos * NCOLS;
    float* o = out + (size_t)pos * HIDDEN;
    const bool has1 = (si >= 1), has2 = (si >= 2);

    for (int c = threadIdx.x; c < HIDDEN; c += 256) {
        const float w0 = conv_w[c * 3 + 0];
        const float w1 = conv_w[c * 3 + 1];
        const float w2 = conv_w[c * 3 + 2];
        const float cur = g0[c];
        const float p1  = has1 ? g0[c - NCOLS]     : 0.0f;
        const float p2  = has2 ? g0[c - 2 * NCOLS] : 0.0f;
        o[c] = cur + w2 * cur + w1 * p1 + w0 * p2;
    }
}

// ---------------------------------------------------------------------------
extern "C" void kernel_launch(void* const* d_in, const int* in_sizes, int n_in,
                              void* d_out, int out_size, void* d_ws, size_t ws_size,
                              hipStream_t stream)
{
    const float*     hidden = (const float*)d_in[0];
    const long long* ids    = (const long long*)d_in[1];
    const float*     tables = (const float*)d_in[2];
    const float*     Wk     = (const float*)d_in[3];
    const float*     Wv     = (const float*)d_in[4];
    const float*     knw    = (const float*)d_in[5];
    const float*     vnw    = (const float*)d_in[6];
    const float*     convw  = (const float*)d_in[7];
    const long long* hm2    = (const long long*)d_in[8];
    const long long* ho2    = (const long long*)d_in[9];
    const long long* hm3    = (const long long*)d_in[10];
    const long long* ho3    = (const long long*)d_in[11];

    float* mkv = (float*)d_ws;           // 16384 x 4096 f32 (256 MB)
    float* out = (float*)d_out;          // 16384 x 2048 f32

    const int POSITIONS = 4 * SEQ;       // 16384
    const int MTILES = POSITIONS / 128;  // 128
    const int NTILES = NCOLS / 64;       // 64

    engram_gemm_kernel<<<MTILES * NTILES, 128, 0, stream>>>(
        ids, tables, Wk, Wv, hm2, ho2, hm3, ho3, mkv);
    engram_gate_kernel<<<POSITIONS, 256, 0, stream>>>(hidden, knw, vnw, mkv);
    engram_conv_kernel<<<POSITIONS, 256, 0, stream>>>(mkv, convw, out);
}
